// Model_39676907885159
// MI455X (gfx1250) — matrix operand order repaired and checked by reading, not compiled, not run on hardware
//
#include <hip/hip_runtime.h>
#include <math.h>

// Flash-attention for B=4,H=16,S=2048,D=64 fp32 with fused (pre-materialized)
// dropout mask. f16 WMMA (v_wmma_f32_16x16x32_f16) with f32 accumulation.
//
// Roofline: dropout mask stream (1.07 GB fp32) dominates -> memory bound
// (~46 us floor at 23.3 TB/s). Mask is read exactly once, coalesced, with
// nontemporal hints so K/V stay hot in the 192MB L2 across the 16 query-tile
// workgroups per (b,h). Softmax cross-lane reductions use DPP row_xmask
// (VALU) instead of ds_bpermute round-trips.

typedef __attribute__((ext_vector_type(16))) _Float16 v16h;
typedef __attribute__((ext_vector_type(8)))  _Float16 v8h;
typedef __attribute__((ext_vector_type(4)))  _Float16 v4h;
typedef __attribute__((ext_vector_type(8)))  float    v8f;

#define B_  4
#define H_  16
#define S_  2048
#define D_  64
#define MT  128          // query rows per workgroup (8 waves x 16 rows)
#define NT  64           // key rows per block iteration
#define LS  72           // LDS row pitch in halves (144B) -> conflict-free b128 frags
#define QK_SCALE 1.4142135623730951f   // 1/sqrt(0.5), folded into f16(Q)
#define L2E      1.4426950408889634f   // log2(e)

// Combine two 8-half (16B) LDS loads into one 16-half A/B fragment.
__device__ __forceinline__ v16h ld_frag16(const _Float16* p0, const _Float16* p1) {
    v8h lo = *(const v8h*)p0;
    v8h hi = *(const v8h*)p1;
    return __builtin_shufflevector(lo, hi,
        0, 1, 2, 3, 4, 5, 6, 7, 8, 9, 10, 11, 12, 13, 14, 15);
}

// XOR-butterfly within each 16-lane row via DPP row_xmask (no LDS, no waits).
// Masks 1/2/4/8 never cross the lane-16 boundary, matching the C-fragment
// half-wave row split.
template <int CTRL>
__device__ __forceinline__ float dpp_xor_f(float x) {
    int v = __builtin_amdgcn_update_dpp(
        __builtin_bit_cast(int, x), __builtin_bit_cast(int, x),
        CTRL, 0xF, 0xF, true);
    return __builtin_bit_cast(float, v);
}

__device__ __forceinline__ float half_row_max(float x) {
    x = fmaxf(x, dpp_xor_f<0x161>(x));   // row_xmask:1
    x = fmaxf(x, dpp_xor_f<0x162>(x));   // row_xmask:2
    x = fmaxf(x, dpp_xor_f<0x164>(x));   // row_xmask:4
    x = fmaxf(x, dpp_xor_f<0x168>(x));   // row_xmask:8
    return x;
}
__device__ __forceinline__ float half_row_sum(float x) {
    x += dpp_xor_f<0x161>(x);
    x += dpp_xor_f<0x162>(x);
    x += dpp_xor_f<0x164>(x);
    x += dpp_xor_f<0x168>(x);
    return x;
}

__global__ __launch_bounds__(256)
void fa_dropout_kernel(const float* __restrict__ q,
                       const float* __restrict__ k,
                       const float* __restrict__ v,
                       const float* __restrict__ mask,
                       float* __restrict__ out) {
    __shared__ __align__(16) _Float16 Qs[MT * LS];       // 18,432 B
    __shared__ __align__(16) _Float16 Ks[NT * LS];       //  9,216 B
    __shared__ __align__(16) _Float16 Vts[D_ * LS];      //  9,216 B  (V transposed: [d][n])
    __shared__ __align__(16) _Float16 Ps[8 * 16 * LS];   // 18,432 B  (per-wave P staging)

    const int tid  = threadIdx.x;
    const int lane = tid & 31;
    const int wave = tid >> 5;
    const int hl   = lane & 15;     // half-lane index (column within 16)
    const int hi   = lane >> 4;     // 0: M rows 0-7 / K lo, 1: M rows 8-15 / K hi

    const int bh = blockIdx.y;                 // fused (b,h)
    const int qb = blockIdx.x * MT;            // query tile base row

    const float* qp = q    + (size_t)bh * S_ * D_;
    const float* kp = k    + (size_t)bh * S_ * D_;
    const float* vp = v    + (size_t)bh * S_ * D_;
    const float* mp = mask + (size_t)bh * S_ * S_;
    float*       op = out  + (size_t)bh * S_ * D_;

    // ---- Stage Q tile (128x64) f32 -> f16 in LDS; fold in the QK scale ----
#pragma unroll
    for (int it = 0; it < 8; ++it) {
        int idx = tid + it * 256;              // 0..2047 float4's
        int r   = idx >> 4;                    // 16 float4 per row
        int c4  = (idx & 15) * 4;
        float4 f = *(const float4*)(qp + (size_t)(qb + r) * D_ + c4);
        v4h h = { (_Float16)(f.x * QK_SCALE), (_Float16)(f.y * QK_SCALE),
                  (_Float16)(f.z * QK_SCALE), (_Float16)(f.w * QK_SCALE) };
        *(v4h*)(&Qs[r * LS + c4]) = h;
    }
    __syncthreads();

    // ---- Each wave keeps its Q A-fragments (16x32 per D-chunk) in VGPRs ----
    v16h qf[2];
    {
        const _Float16* qrow = &Qs[(wave * 16 + hl) * LS];
#pragma unroll
        for (int c = 0; c < 2; ++c) {
            int d0 = c * 32 + hi * 8;          // A layout: K 0-7 / 8-15 per half-wave
            qf[c] = ld_frag16(qrow + d0, qrow + d0 + 16);
        }
    }

    // Online-softmax state. Element i of each v8f maps to row (hi*8 + i).
    // m_run: running max (raw);  m2_run: running max in log2 domain.
    v8f of[4];                                  // O accum: 16x64 in 4 C-fragments
    v8f m_run, m2_run, l_run;
#pragma unroll
    for (int i = 0; i < 8; ++i) { m_run[i] = -INFINITY; m2_run[i] = -INFINITY; l_run[i] = 0.0f; }
#pragma unroll
    for (int d = 0; d < 4; ++d)
#pragma unroll
        for (int i = 0; i < 8; ++i) of[d][i] = 0.0f;

    _Float16* pw = &Ps[wave * 16 * LS];        // this wave's private P staging

#pragma unroll 1
    for (int kb = 0; kb < S_; kb += NT) {
        __syncthreads();   // previous iteration's K/V reads complete

        // ---- Stage K block (64x64, row-major f16) ----
#pragma unroll
        for (int it = 0; it < 4; ++it) {
            int idx = tid + it * 256;          // 0..1023 float4's
            int r   = idx >> 4;
            int c4  = (idx & 15) * 4;
            float4 f = *(const float4*)(kp + (size_t)(kb + r) * D_ + c4);
            v4h h = { (_Float16)f.x, (_Float16)f.y, (_Float16)f.z, (_Float16)f.w };
            *(v4h*)(&Ks[r * LS + c4]) = h;
        }
        // ---- Stage V block transposed: Vts[d][n] (f16) ----
#pragma unroll
        for (int it = 0; it < 4; ++it) {
            int idx = tid + it * 256;
            int r   = idx >> 4;                // key row n
            int c4  = (idx & 15) * 4;          // d base
            float4 f = *(const float4*)(vp + (size_t)(kb + r) * D_ + c4);
            Vts[(c4 + 0) * LS + r] = (_Float16)f.x;
            Vts[(c4 + 1) * LS + r] = (_Float16)f.y;
            Vts[(c4 + 2) * LS + r] = (_Float16)f.z;
            Vts[(c4 + 3) * LS + r] = (_Float16)f.w;
        }
        __syncthreads();

        // ---- Scores S = Q K^T (scale pre-folded): 4 subtiles x 2 K-chunks ----
        v8f sf[4];
#pragma unroll
        for (int ns = 0; ns < 4; ++ns) {
            v8f acc = {};
            const _Float16* krow = &Ks[(ns * 16 + hl) * LS];  // B col n = ns*16+hl
#pragma unroll
            for (int c = 0; c < 2; ++c) {
                int d0 = c * 32 + hi * 8;      // B layout: K 0-7 / 8-15 per half-wave
                v16h bf = ld_frag16(krow + d0, krow + d0 + 16);
                acc = __builtin_amdgcn_wmma_f32_16x16x32_f16(
                    false, qf[c], false, bf, (short)0, acc, false, false);
            }
            sf[ns] = acc;
        }

        // ---- Row max: 4 frags elementwise, then DPP butterfly over 16 lanes ----
        v8f mx;
#pragma unroll
        for (int i = 0; i < 8; ++i)
            mx[i] = fmaxf(fmaxf(sf[0][i], sf[1][i]), fmaxf(sf[2][i], sf[3][i]));
#pragma unroll
        for (int i = 0; i < 8; ++i)
            mx[i] = half_row_max(mx[i]);

        v8f mn2, alpha;
#pragma unroll
        for (int i = 0; i < 8; ++i) {
            float m_new = fmaxf(m_run[i], mx[i]);
            mn2[i]   = m_new * L2E;
            alpha[i] = __builtin_amdgcn_exp2f(m2_run[i] - mn2[i]); // 0 on first block
            m_run[i] = m_new;
            m2_run[i] = mn2[i];
        }
#pragma unroll
        for (int d = 0; d < 4; ++d)
#pragma unroll
            for (int i = 0; i < 8; ++i) of[d][i] *= alpha[i];

        // ---- P = exp2(S*log2e - m2); l accumulates WITHOUT dropout mask ----
        v8f rs = {};
#pragma unroll
        for (int ns = 0; ns < 4; ++ns)
#pragma unroll
            for (int i = 0; i < 8; ++i) {
                float p = __builtin_amdgcn_exp2f(fmaf(sf[ns][i], L2E, -mn2[i]));
                sf[ns][i] = p;
                rs[i] += p;
            }
#pragma unroll
        for (int i = 0; i < 8; ++i)
            rs[i] = half_row_sum(rs[i]);
#pragma unroll
        for (int i = 0; i < 8; ++i) l_run[i] = l_run[i] * alpha[i] + rs[i];

        // ---- Apply dropout mask (streamed once, nontemporal) and stage P ----
        const float* mrow = mp + (size_t)(qb + wave * 16 + hi * 8) * S_ + kb;
#pragma unroll
        for (int ns = 0; ns < 4; ++ns)
#pragma unroll
            for (int i = 0; i < 8; ++i) {
                float mv = __builtin_nontemporal_load(
                    mrow + (size_t)i * S_ + ns * 16 + hl);
                pw[(hi * 8 + i) * LS + ns * 16 + hl] = (_Float16)(sf[ns][i] * mv);
            }

        // ---- O += P V : re-read P as A-fragments, Vts as B-fragments ----
        v16h pf[2];
        {
            const _Float16* prow = &pw[hl * LS];
#pragma unroll
            for (int c = 0; c < 2; ++c) {
                int n0 = c * 32 + hi * 8;
                pf[c] = ld_frag16(prow + n0, prow + n0 + 16);
            }
        }
#pragma unroll
        for (int ds = 0; ds < 4; ++ds) {
            const _Float16* vrow = &Vts[(ds * 16 + hl) * LS]; // B col d = ds*16+hl
#pragma unroll
            for (int c = 0; c < 2; ++c) {
                int n0 = c * 32 + hi * 8;
                v16h bf = ld_frag16(vrow + n0, vrow + n0 + 16);
                of[ds] = __builtin_amdgcn_wmma_f32_16x16x32_f16(
                    false, pf[c], false, bf, (short)0, of[ds], false, false);
            }
        }
    }

    // ---- Normalize by softmax denominator and store (coalesced, NT) ----
    v8f inv;
#pragma unroll
    for (int i = 0; i < 8; ++i) inv[i] = 1.0f / l_run[i];

    float* orow = op + (size_t)(qb + wave * 16 + hi * 8) * D_;
#pragma unroll
    for (int ds = 0; ds < 4; ++ds)
#pragma unroll
        for (int i = 0; i < 8; ++i)
            __builtin_nontemporal_store(of[ds][i] * inv[i],
                                        orow + (size_t)i * D_ + ds * 16 + hl);
}

extern "C" void kernel_launch(void* const* d_in, const int* in_sizes, int n_in,
                              void* d_out, int out_size, void* d_ws, size_t ws_size,
                              hipStream_t stream) {
    const float* q    = (const float*)d_in[0];
    const float* k    = (const float*)d_in[1];
    const float* v    = (const float*)d_in[2];
    const float* mask = (const float*)d_in[3];
    float* out = (float*)d_out;

    dim3 grid(S_ / MT, B_ * H_);   // (16, 64)
    dim3 block(256);               // 8 wave32
    fa_dropout_kernel<<<grid, block, 0, stream>>>(q, k, v, mask, out);
}
